// ScaledDotProductAttention_73942156968659
// MI455X (gfx1250) — matrix operand order repaired and checked by reading, not compiled, not run on hardware
//
#include <hip/hip_runtime.h>

// Flash attention forward (causal), fp32 in/out. MI455X (gfx1250) version:
//  - pre-pass converts K -> f16 row-major and V -> f16 transposed in d_ws
//  - main kernel stages K/V tiles into LDS with the Tensor Data Mover
//    (tensor_load_to_lds, TENSORcnt), double-buffered to overlap DMA with
//    v_wmma_f32_16x16x32_f16 compute (fp32 accumulate)
// B=2, H=16, S=2048, D=64. Mask input is causal tril -> applied analytically.
// Workspace requirement: 2 * 32*2048*64*2 bytes = 16 MiB (Kh + Vt).

typedef __attribute__((ext_vector_type(16))) _Float16     v16h;
typedef __attribute__((ext_vector_type(8)))  float        v8f;
typedef __attribute__((ext_vector_type(4)))  unsigned int u32x4;
typedef __attribute__((ext_vector_type(8)))  int          i32x8;
typedef __attribute__((ext_vector_type(4)))  int          i32x4;

#define SLEN 2048
#define DH   64
#define QT   128     // query rows per workgroup (8 waves x 16 rows)
#define KB   32      // key block size per iteration
#define KP   72      // LDS pitch (halves) for K tile rows   (144B = 128B + 16B pad)
#define VP   40      // LDS pitch (halves) for V^T tile rows (80B  =  64B + 16B pad)
#define PP   40      // LDS pitch (halves) for P scratch rows

// ---------------- pre-pass: f32 -> f16, and transpose V ----------------
__global__ __launch_bounds__(256, 1)
void cvt_kv(const float* __restrict__ Kg, const float* __restrict__ Vg,
            _Float16* __restrict__ Kh, _Float16* __restrict__ Vt)
{
    __shared__ _Float16 tile[64 * 72];          // V sub-tile, row-major staging

    const int bh  = blockIdx.y;                 // 0..31
    const int s0  = blockIdx.x * 64;            // sequence tile
    const int row = threadIdx.x >> 2;           // 0..63
    const int seg = (threadIdx.x & 3) * 16;     // 0,16,32,48
    const size_t gbase = (size_t)bh * (SLEN * DH);

    {   // K: straight convert, coalesced in/out
        const float*  src = Kg + gbase + (size_t)(s0 + row) * DH + seg;
        _Float16*     dst = Kh + gbase + (size_t)(s0 + row) * DH + seg;
        #pragma unroll
        for (int e = 0; e < 16; ++e) dst[e] = (_Float16)src[e];
    }
    {   // V: stage rows as f16 in LDS
        const float* src = Vg + gbase + (size_t)(s0 + row) * DH + seg;
        #pragma unroll
        for (int e = 0; e < 16; ++e) tile[row * 72 + seg + e] = (_Float16)src[e];
    }
    __syncthreads();
    {   // write transposed: Vt[bh][d][s], coalesced 32B stores per thread
        _Float16* dst = Vt + gbase + (size_t)row * SLEN + s0 + seg;
        #pragma unroll
        for (int e = 0; e < 16; ++e) dst[e] = tile[(seg + e) * 72 + row];
    }
}

// ---------------- TDM descriptor issue (2D tile, LDS row padding) ----------------
__device__ __forceinline__ void tdm_load_2d(unsigned lds_addr, unsigned long long gaddr,
                                            unsigned dim0, unsigned dim1,
                                            unsigned tile0, unsigned tile1,
                                            unsigned long long stride0,
                                            unsigned pad_interval, unsigned pad_amount)
{
    u32x4 g0;
    g0[0] = 1u;                                              // count=1, user-mode load
    g0[1] = lds_addr;                                        // LDS byte address
    g0[2] = (unsigned)gaddr;                                 // global_addr[31:0]
    g0[3] = ((unsigned)(gaddr >> 32) & 0x01FFFFFFu)          // global_addr[56:32]
            | (2u << 30);                                    // type = 2 ("image")
    i32x8 g1;
    g1[0] = (int)((1u << 16)                                 // data_size = 2 bytes
                | (1u << 20)                                 // pad_enable
                | (pad_interval << 22) | (pad_amount << 25));
    g1[1] = (int)((dim0 & 0xFFFFu) << 16);                   // tensor_dim0[15:0]
    g1[2] = (int)((dim0 >> 16) | ((dim1 & 0xFFFFu) << 16));  // dim0[31:16] | dim1[15:0]
    g1[3] = (int)((dim1 >> 16) | (tile0 << 16));             // dim1[31:16] | tile_dim0
    g1[4] = (int)(tile1 & 0xFFFFu);                          // tile_dim1 (tile_dim2=0)
    g1[5] = (int)(unsigned)stride0;                          // tensor_dim0_stride[31:0]
    g1[6] = (int)((unsigned)(stride0 >> 32) & 0xFFFFu);      // stride[47:32]
    g1[7] = 0;
    i32x4 z4 = {};                                           // 2D: groups 2/3 unused
    i32x8 z8 = {};
    __builtin_amdgcn_tensor_load_to_lds(g0, g1, z4, z4, z8, 0);
}

// ---------------- main flash-attention kernel ----------------
__global__ __launch_bounds__(256, 1)
void fa_fwd_causal(const float* __restrict__ Qg,
                   const _Float16* __restrict__ Kh,   // [bh][s][64] f16
                   const _Float16* __restrict__ Vt,   // [bh][64][s] f16
                   float* __restrict__ Og)
{
    __shared__ _Float16 sK [2][KB * KP];     // K tile,  row-major [key][d], padded
    __shared__ _Float16 sVt[2][DH * VP];     // V tile,  transposed [d][key], padded
    __shared__ _Float16 sP [8 * 16 * PP];    // per-wave P scratch

    const int tid   = threadIdx.x;
    const int lane  = tid & 31;
    const int w     = tid >> 5;              // wave id 0..7
    const int hi    = lane >> 4;             // half-wave
    const int n     = lane & 15;
    const int abase = hi * 8;                // A-layout K base
    const int koff  = hi * 16;               // K base for A and B (contiguous k = 16*hi + s)

    const size_t bh_base = (size_t)blockIdx.y * (size_t)(SLEN * DH);
    const int q0 = blockIdx.x * QT + w * 16;

    const _Float16* khBH = Kh + bh_base;
    const _Float16* vtBH = Vt + bh_base;
    const int jmax = blockIdx.x * 4 + 3;

    // ---- Q tile (16 x 64) -> two f16 A-operands ----
    v16h qa0, qa1;
    {
        const float* qrow = Qg + bh_base + (size_t)(q0 + n) * DH;
        #pragma unroll
        for (int h = 0; h < 8; ++h) {
            qa0[h]     = (_Float16)qrow[koff + h];
            qa0[h + 8] = (_Float16)qrow[koff + 8 + h];
            qa1[h]     = (_Float16)qrow[32 + koff + h];
            qa1[h + 8] = (_Float16)qrow[32 + koff + 8 + h];
        }
    }

    v8f o0 = {}, o1 = {}, o2 = {}, o3 = {};
    float m[8], l[8];
    #pragma unroll
    for (int i = 0; i < 8; ++i) { m[i] = -__builtin_inff(); l[i] = 0.0f; }
    const float scale = 0.125f;

    // ---- TDM issue for one key block (wave 0 only; EXEC-independent DMA) ----
    auto issue_block = [&](int j, int buf) {
        // K tile: 32 rows x 64 halves, src stride 64, LDS pad 16B per 128B row
        tdm_load_2d((unsigned)(size_t)&sK[buf][0],
                    (unsigned long long)(size_t)(khBH + (size_t)(j * KB) * DH),
                    /*dim0*/ DH, /*dim1*/ SLEN, /*tile0*/ DH, /*tile1*/ KB,
                    /*stride0*/ DH, /*pad_interval(128B)*/ 4, /*pad_amount(4DW)*/ 3);
        // V^T tile: 64 rows x 32 halves, src stride 2048, LDS pad 16B per 64B row
        tdm_load_2d((unsigned)(size_t)&sVt[buf][0],
                    (unsigned long long)(size_t)(vtBH + (size_t)(j * KB)),
                    /*dim0*/ SLEN, /*dim1*/ DH, /*tile0*/ KB, /*tile1*/ DH,
                    /*stride0*/ SLEN, /*pad_interval(64B)*/ 3, /*pad_amount(4DW)*/ 3);
    };

    if (w == 0) issue_block(0, 0);

    for (int j = 0; j <= jmax; ++j) {
        const int buf = j & 1;
        if (w == 0) {
            if (j < jmax) {                       // prefetch next block, then wait
                issue_block(j + 1, buf ^ 1);      // for block j (TDM is in-order)
                __builtin_amdgcn_s_wait_tensorcnt(2);
            } else {
                __builtin_amdgcn_s_wait_tensorcnt(0);
            }
        }
        __syncthreads();                          // block-j tiles visible to all waves

        if (KB * j <= q0 + 15) {                  // causally live for this wave
            // ---- scores S = (Q K^T) * scale : two 16x16 tiles ----
            v8f s1, s2;
            #pragma unroll
            for (int sub = 0; sub < 2; ++sub) {
                const _Float16* kr = &sK[buf][(sub * 16 + n) * KP + koff];
                v16h b0, b1;
                #pragma unroll
                for (int h = 0; h < 16; ++h) { b0[h] = kr[h]; b1[h] = kr[32 + h]; }
                v8f s = {};
                s = __builtin_amdgcn_wmma_f32_16x16x32_f16(false, qa0, false, b0,
                                                           (short)0, s, false, false);
                s = __builtin_amdgcn_wmma_f32_16x16x32_f16(false, qa1, false, b1,
                                                           (short)0, s, false, false);
                if (sub == 0) s1 = s; else s2 = s;
            }

            // ---- causal mask + scale; C frag: elem i -> row i+8*hi, col n ----
            const int colbase = j * KB;
            #pragma unroll
            for (int i = 0; i < 8; ++i) {
                const int qrow = q0 + i + 8 * hi;
                float a = s1[i] * scale;
                float b = s2[i] * scale;
                if (colbase + n      > qrow) a = -__builtin_inff();
                if (colbase + 16 + n > qrow) b = -__builtin_inff();
                s1[i] = a; s2[i] = b;
            }

            // ---- online softmax update ----
            #pragma unroll
            for (int i = 0; i < 8; ++i) {
                float rm = fmaxf(s1[i], s2[i]);
                #pragma unroll
                for (int d = 1; d < 16; d <<= 1) rm = fmaxf(rm, __shfl_xor(rm, d, 32));
                const float mn    = fmaxf(m[i], rm);
                const float alpha = __expf(m[i] - mn);
                const float p1 = __expf(s1[i] - mn);
                const float p2 = __expf(s2[i] - mn);
                float rs = p1 + p2;
                #pragma unroll
                for (int d = 1; d < 16; d <<= 1) rs += __shfl_xor(rs, d, 32);
                l[i] = l[i] * alpha + rs;
                m[i] = mn;
                o0[i] *= alpha; o1[i] *= alpha; o2[i] *= alpha; o3[i] *= alpha;
                s1[i] = p1; s2[i] = p2;
            }

            // ---- C-layout P -> A-layout f16 via per-wave LDS scratch ----
            _Float16* pw = &sP[w * 16 * PP];
            #pragma unroll
            for (int i = 0; i < 8; ++i) {
                const int row = i + 8 * hi;
                pw[row * PP + n]      = (_Float16)s1[i];
                pw[row * PP + 16 + n] = (_Float16)s2[i];
            }
            v16h pa;
            {
                const _Float16* pr = &pw[n * PP];
                #pragma unroll
                for (int h = 0; h < 8; ++h) {
                    pa[h]     = pr[koff + h];
                    pa[h + 8] = pr[koff + 8 + h];
                }
            }

            // ---- O += P V : 4 tiles along D ----
            #pragma unroll
            for (int t = 0; t < 4; ++t) {
                const _Float16* vr = &sVt[buf][(t * 16 + n) * VP + koff];
                v16h vb;
                #pragma unroll
                for (int h = 0; h < 16; ++h) vb[h] = vr[h];
                v8f acc = (t == 0) ? o0 : (t == 1) ? o1 : (t == 2) ? o2 : o3;
                acc = __builtin_amdgcn_wmma_f32_16x16x32_f16(false, pa, false, vb,
                                                             (short)0, acc, false, false);
                if (t == 0) o0 = acc; else if (t == 1) o1 = acc;
                else if (t == 2) o2 = acc; else o3 = acc;
            }
        }
        __syncthreads();   // all waves done with buf before it is DMA-overwritten
    }

    // ---- epilogue: O / l ----
    #pragma unroll
    for (int i = 0; i < 8; ++i) {
        const int row = q0 + i + 8 * hi;
        const float inv = 1.0f / l[i];
        float* orow = Og + bh_base + (size_t)row * DH;
        orow[n]      = o0[i] * inv;
        orow[16 + n] = o1[i] * inv;
        orow[32 + n] = o2[i] * inv;
        orow[48 + n] = o3[i] * inv;
    }
}

extern "C" void kernel_launch(void* const* d_in, const int* in_sizes, int n_in,
                              void* d_out, int out_size, void* d_ws, size_t ws_size,
                              hipStream_t stream) {
    (void)in_sizes; (void)n_in; (void)out_size; (void)ws_size;
    const float* Q = (const float*)d_in[0];
    const float* K = (const float*)d_in[1];
    const float* V = (const float*)d_in[2];
    // d_in[3] is the causal mask; applied analytically in-kernel.
    float* O = (float*)d_out;

    const int B = 2, H = 16;
    _Float16* Kh = (_Float16*)d_ws;                          // 8 MiB
    _Float16* Vt = Kh + (size_t)(B * H) * SLEN * DH;         // 8 MiB

    cvt_kv<<<dim3(SLEN / 64, B * H), 256, 0, stream>>>(K, V, Kh, Vt);
    fa_fwd_causal<<<dim3(SLEN / QT, B * H), 256, 0, stream>>>(Q, Kh, Vt, O);
}
